// CMAE_72894184947729
// MI455X (gfx1250) — matrix operand order repaired and checked by reading, not compiled, not run on hardware
//
#include <hip/hip_runtime.h>
#include <hip/hip_bf16.h>

#define NN 50000
#define EE 800000
#define BB 256
#define DD 128
#define HH 256
#define LL 3
#define NMASK 25000
#define BN_EPS 1e-5f
#define TEMP 0.2f

typedef _Float16 v16h __attribute__((ext_vector_type(16)));
typedef float    v8f  __attribute__((ext_vector_type(8)));
typedef unsigned int v4u __attribute__((ext_vector_type(4)));

union Frag16 { v16h h; v4u u[2]; };

// ---------------------------------------------------------------------------
// WMMA GEMM: C[M x Nc] = A[M x K] * B[K x Nc], A row-major f16,
// B supplied TRANSPOSED as Bt[Nc x K] f16 (so both fragments load contiguous),
// C row-major f32. M % 16 == 0, K % 32 == 0, Nc % 16 == 0.
// One wave computes one 16x16 C tile, K-loop in steps of 32 via
// v_wmma_f32_16x16x32_f16 (fp32 accumulate).
// ---------------------------------------------------------------------------
__global__ void __launch_bounds__(256)
wmma_gemm_f16(const _Float16* __restrict__ A, const _Float16* __restrict__ Bt,
              float* __restrict__ C, int M, int K, int Nc) {
  int wave = blockIdx.x * 8 + (threadIdx.x >> 5);
  int lane = threadIdx.x & 31;
  int tilesN = Nc >> 4;
  int numTiles = (M >> 4) * tilesN;
  if (wave >= numTiles) return;           // wave-uniform: EXEC stays all-ones
  int tm = wave / tilesN;
  int tn = wave % tilesN;
  int r  = lane & 15;
  int hs = lane >> 4;

  const _Float16* arow = A  + (size_t)(tm * 16 + r) * K;
  const _Float16* brow = Bt + (size_t)(tn * 16 + r) * K;

  v8f acc = {};
  for (int k = 0; k < K; k += 32) {
    Frag16 a, b;
    // A 16x32 f16 fragment: j<8 -> K = k + hs*8 + j ; j>=8 -> K = k+16+hs*8+(j-8)
    a.u[0] = *(const v4u*)(arow + k + hs * 8);
    a.u[1] = *(const v4u*)(arow + k + 16 + hs * 8);
    // B 32x16 f16 fragment (Bt row = column n): j<8 -> K = k + hs*8 + j ; j>=8 -> K = k+16+hs*8+(j-8)
    b.u[0] = *(const v4u*)(brow + k + hs * 8);
    b.u[1] = *(const v4u*)(brow + k + 16 + hs * 8);
    acc = __builtin_amdgcn_wmma_f32_16x16x32_f16(false, a.h, false, b.h,
                                                 (short)0, acc, false, false);
  }
  // C/D layout: VGPR v -> row tm*16 + v + 8*hs, col tn*16 + r
  float* crow = C + (size_t)(tm * 16 + hs * 8) * Nc + tn * 16 + r;
#pragma unroll
  for (int v = 0; v < 8; ++v) crow[(size_t)v * Nc] = acc[v];
}

// --- transpose-convert weights: W[Lc][K][O] f32 -> out[Lc][O][K] f16 --------
__global__ void cvt_wT(const float* __restrict__ W, _Float16* __restrict__ out,
                       int Lc, int K, int O) {
  int t = blockIdx.x * blockDim.x + threadIdx.x;
  int total = Lc * K * O;
  if (t >= total) return;
  int l = t / (K * O);
  int k = (t / O) % K;
  int o = t % O;
  out[((size_t)l * O + o) * K + k] = (_Float16)W[t];
}

// --- masking: x[mask[k]][:] = tok[:] ----------------------------------------
__global__ void apply_mask(float* __restrict__ x, const int* __restrict__ mask,
                           const float* __restrict__ tok) {
  int t = blockIdx.x * blockDim.x + threadIdx.x;
  if (t >= NMASK * DD) return;
  int k = t >> 7, d = t & 127;
  x[(size_t)mask[k] * DD + d] = tok[d];
}

// --- edge scatter: agg[dst[e]] += h[src[e]]  (float4 granularity) ----------
__global__ void edge_scatter(const float* __restrict__ h, const int* __restrict__ src,
                             const int* __restrict__ dst, float* __restrict__ agg) {
  int t = blockIdx.x * blockDim.x + threadIdx.x;
  if (t >= EE * (DD / 4)) return;
  int e = t >> 5, c = t & 31;
  float4 v = *((const float4*)(h + (size_t)src[e] * DD) + c);
  float* o = agg + (size_t)dst[e] * DD + c * 4;
  atomicAdd(o + 0, v.x); atomicAdd(o + 1, v.y);
  atomicAdd(o + 2, v.z); atomicAdd(o + 3, v.w);
}

// --- (h + agg) -> f16 -------------------------------------------------------
__global__ void add_cvt(const float* __restrict__ h, const float* __restrict__ agg,
                        _Float16* __restrict__ out, int total) {
  int t = blockIdx.x * blockDim.x + threadIdx.x;
  if (t >= total) return;
  out[t] = (_Float16)(h[t] + agg[t]);
}

// --- BN column reduce: 128 rows per block, atomics once per block ----------
__global__ void bn_reduce(const float* __restrict__ z, float* __restrict__ sums,
                          float* __restrict__ sqs, int Nrows, int C) {
  int tpc = blockDim.x / C;          // 1 (C=256) or 2 (C=128)
  int c   = threadIdx.x % C;
  int sub = threadIdx.x / C;
  int r0 = blockIdx.x * 128;
  int r1 = r0 + 128; if (r1 > Nrows) r1 = Nrows;
  float s = 0.f, q = 0.f;
  for (int r = r0 + sub; r < r1; r += tpc) {
    float v = z[(size_t)r * C + c];
    s += v; q += v * v;
  }
  atomicAdd(&sums[c], s);
  atomicAdd(&sqs[c], q);
}

__global__ void bn_final(const float* __restrict__ sums, const float* __restrict__ sqs,
                         const float* __restrict__ g, const float* __restrict__ b,
                         float* __restrict__ cA, float* __restrict__ cB,
                         int Nrows, int C) {
  int c = blockIdx.x * blockDim.x + threadIdx.x;
  if (c >= C) return;
  float inv_n = 1.0f / (float)Nrows;
  float mean = sums[c] * inv_n;
  float var  = sqs[c] * inv_n - mean * mean;
  float a = g[c] * rsqrtf(var + BN_EPS);
  cA[c] = a;
  cB[c] = b[c] - a * mean;
}

__global__ void bn_apply_h16(const float* __restrict__ z, const float* __restrict__ cA,
                             const float* __restrict__ cB, _Float16* __restrict__ out,
                             int total, int C) {
  int t = blockIdx.x * blockDim.x + threadIdx.x;
  if (t >= total) return;
  int c = t % C;
  float r = fmaf(cA[c], z[t], cB[c]);
  out[t] = (_Float16)fmaxf(r, 0.f);
}

__global__ void bn_apply_f32(const float* __restrict__ z, const float* __restrict__ cA,
                             const float* __restrict__ cB, float* __restrict__ out,
                             int total, int C) {
  int t = blockIdx.x * blockDim.x + threadIdx.x;
  if (t >= total) return;
  int c = t % C;
  out[t] = fmaxf(fmaf(cA[c], z[t], cB[c]), 0.f);
}

// --- graph pooling: pooled[gid[n]][layer*DD + d] += h[n][d] ----------------
__global__ void pool_sum(const float* __restrict__ h, const int* __restrict__ gid,
                         float* __restrict__ pooled, int layer) {
  int t = blockIdx.x * blockDim.x + threadIdx.x;
  if (t >= NN * DD) return;
  int n = t >> 7, d = t & 127;
  atomicAdd(&pooled[(size_t)gid[n] * (LL * DD) + layer * DD + d], h[t]);
}

// --- projection head (tiny fp32 GEMMs) -------------------------------------
__global__ void proj1_k(const float* __restrict__ g, const float* __restrict__ W,
                        const float* __restrict__ bias, float* __restrict__ out) {
  int t = blockIdx.x * blockDim.x + threadIdx.x;
  if (t >= BB * 192) return;
  int i = t / 192, j = t % 192;
  float s = bias[j];
  const float* gr = g + (size_t)i * 384;
  for (int k = 0; k < 384; ++k) s += gr[k] * W[(size_t)k * 192 + j];
  out[t] = fmaxf(s, 0.f);
}

__global__ void proj2_k(const float* __restrict__ h, const float* __restrict__ W,
                        const float* __restrict__ bias, float* __restrict__ out) {
  int t = blockIdx.x * blockDim.x + threadIdx.x;
  if (t >= BB * 128) return;
  int i = t / 128, j = t % 128;
  float s = bias[j];
  const float* hr = h + (size_t)i * 192;
  for (int k = 0; k < 192; ++k) s += hr[k] * W[(size_t)k * 128 + j];
  out[t] = s;
}

// --- loss -------------------------------------------------------------------
__global__ void row_norms(const float* __restrict__ x, const float* __restrict__ y,
                          float* __restrict__ norms) {
  __shared__ float red[128];
  int row = blockIdx.x;   // 0..255 -> x, 256..511 -> y
  const float* p = (row < BB) ? (x + (size_t)row * 128)
                              : (y + (size_t)(row - BB) * 128);
  float v = p[threadIdx.x];
  red[threadIdx.x] = v * v;
  __syncthreads();
  for (int s = 64; s > 0; s >>= 1) {
    if (threadIdx.x < s) red[threadIdx.x] += red[threadIdx.x + s];
    __syncthreads();
  }
  if (threadIdx.x == 0) norms[row] = sqrtf(red[0]);
}

__global__ void nt_xent(const float* __restrict__ X, const float* __restrict__ Y,
                        const float* __restrict__ norms, float* __restrict__ out) {
  __shared__ float xi[128];
  __shared__ float ev[256];
  __shared__ float red[256];
  int i = blockIdx.x, t = threadIdx.x;
  if (t < 128) xi[t] = X[(size_t)i * 128 + t];
  __syncthreads();
  float dot = 0.f;
  const float* yr = Y + (size_t)t * 128;
#pragma unroll 4
  for (int k = 0; k < 128; ++k) dot += xi[k] * yr[k];
  float sim = dot / (norms[i] * norms[BB + t]);
  float e = expf(sim * (1.0f / TEMP));
  ev[t] = e; red[t] = e;
  __syncthreads();
  for (int s = 128; s > 0; s >>= 1) {
    if (t < s) red[t] += red[t + s];
    __syncthreads();
  }
  if (t == 0) {
    float sum = red[0];
    float pos = ev[i];
    atomicAdd(out, -logf(pos / (sum - pos)) * (1.0f / (float)BB));
  }
}

// ===========================================================================
// Host side
// ===========================================================================
static inline int cdiv(long long a, long long b) { return (int)((a + b - 1) / b); }

static void run_encoder(const float* h0, const int* src, const int* dst,
                        const int* gid,
                        const _Float16* w1hT, const float* g1, const float* b1,
                        const _Float16* w2hT, const float* g2, const float* b2,
                        float* pooled,
                        float* h, float* agg /* also z2 */, float* z1,
                        _Float16* ah16, _Float16* z1h,
                        float* sums, float* sqs, float* cA, float* cB,
                        hipStream_t stream) {
  const int TB = 256;
  for (int i = 0; i < LL; ++i) {
    const float* hin = (i == 0) ? h0 : h;
    // aggregate neighbors
    hipMemsetAsync(agg, 0, (size_t)NN * DD * sizeof(float), stream);
    edge_scatter<<<cdiv((long long)EE * (DD / 4), TB), TB, 0, stream>>>(hin, src, dst, agg);
    // (h + agg) -> f16
    add_cvt<<<cdiv((long long)NN * DD, TB), TB, 0, stream>>>(hin, agg, ah16, NN * DD);
    // GEMM1: [N,128] x [128,256]
    wmma_gemm_f16<<<cdiv((long long)(NN / 16) * (HH / 16), 8), TB, 0, stream>>>(
        ah16, w1hT + (size_t)i * HH * DD, z1, NN, DD, HH);
    // BN1 + ReLU -> f16
    hipMemsetAsync(sums, 0, HH * sizeof(float), stream);
    hipMemsetAsync(sqs, 0, HH * sizeof(float), stream);
    bn_reduce<<<cdiv(NN, 128), TB, 0, stream>>>(z1, sums, sqs, NN, HH);
    bn_final<<<1, HH, 0, stream>>>(sums, sqs, g1 + i * HH, b1 + i * HH, cA, cB, NN, HH);
    bn_apply_h16<<<cdiv((long long)NN * HH, TB), TB, 0, stream>>>(z1, cA, cB, z1h, NN * HH, HH);
    // GEMM2: [N,256] x [256,128] -> z2 (aliases agg, dead by now)
    wmma_gemm_f16<<<cdiv((long long)(NN / 16) * (DD / 16), 8), TB, 0, stream>>>(
        z1h, w2hT + (size_t)i * DD * HH, agg, NN, HH, DD);
    // BN2 + ReLU -> f32 h
    hipMemsetAsync(sums, 0, DD * sizeof(float), stream);
    hipMemsetAsync(sqs, 0, DD * sizeof(float), stream);
    bn_reduce<<<cdiv(NN, 128), TB, 0, stream>>>(agg, sums, sqs, NN, DD);
    bn_final<<<1, DD, 0, stream>>>(sums, sqs, g2 + i * DD, b2 + i * DD, cA, cB, NN, DD);
    bn_apply_f32<<<cdiv((long long)NN * DD, TB), TB, 0, stream>>>(agg, cA, cB, h, NN * DD, DD);
    // pool
    pool_sum<<<cdiv((long long)NN * DD, TB), TB, 0, stream>>>(h, gid, pooled, i);
  }
}

extern "C" void kernel_launch(void* const* d_in, const int* in_sizes, int n_in,
                              void* d_out, int out_size, void* d_ws, size_t ws_size,
                              hipStream_t stream) {
  (void)in_sizes; (void)n_in; (void)out_size; (void)ws_size;
  const float* feat       = (const float*)d_in[0];
  const int*   src        = (const int*)d_in[1];
  const int*   dst        = (const int*)d_in[2];
  const int*   gid        = (const int*)d_in[3];
  const int*   mask_nodes = (const int*)d_in[4];
  const float* mask_token = (const float*)d_in[5];
  const float* enc_W1 = (const float*)d_in[6];
  const float* enc_g1 = (const float*)d_in[7];
  const float* enc_b1 = (const float*)d_in[8];
  const float* enc_W2 = (const float*)d_in[9];
  const float* enc_g2 = (const float*)d_in[10];
  const float* enc_b2 = (const float*)d_in[11];
  const float* con_W1 = (const float*)d_in[12];
  const float* con_g1 = (const float*)d_in[13];
  const float* con_b1 = (const float*)d_in[14];
  const float* con_W2 = (const float*)d_in[15];
  const float* con_g2 = (const float*)d_in[16];
  const float* con_b2 = (const float*)d_in[17];
  const float* pW1 = (const float*)d_in[18];
  const float* pb1 = (const float*)d_in[19];
  const float* pW2 = (const float*)d_in[20];
  const float* pb2 = (const float*)d_in[21];
  float* out = (float*)d_out;

  // workspace carve-up
  char* base = (char*)d_ws;
  size_t off = 0;
  auto carve = [&](size_t bytes) -> void* {
    off = (off + 255) & ~(size_t)255;
    void* p = base + off;
    off += bytes;
    return p;
  };
  float*    buf_x   = (float*)carve((size_t)NN * DD * 4);
  float*    buf_h   = (float*)carve((size_t)NN * DD * 4);
  float*    buf_agg = (float*)carve((size_t)NN * DD * 4);   // doubles as z2
  float*    buf_z1  = (float*)carve((size_t)NN * HH * 4);
  _Float16* ah16    = (_Float16*)carve((size_t)NN * DD * 2);
  _Float16* z1h     = (_Float16*)carve((size_t)NN * HH * 2);
  _Float16* w1hT    = (_Float16*)carve((size_t)2 * LL * HH * DD * 2); // [enc][L][H][D]
  _Float16* w2hT    = (_Float16*)carve((size_t)2 * LL * DD * HH * 2); // [enc][L][D][H]
  float*    sums    = (float*)carve(HH * 4);
  float*    sqs     = (float*)carve(HH * 4);
  float*    cA      = (float*)carve(HH * 4);
  float*    cB      = (float*)carve(HH * 4);
  float*    pooled_g = (float*)carve((size_t)BB * LL * DD * 4); // gh (masked/online)
  float*    pooled_c = (float*)carve((size_t)BB * LL * DD * 4); // ch (contrast)
  float*    hid_h   = (float*)carve((size_t)BB * 192 * 4);
  float*    hid_m   = (float*)carve((size_t)BB * 192 * 4);
  float*    c_h     = (float*)carve((size_t)BB * 128 * 4);
  float*    c_m     = (float*)carve((size_t)BB * 128 * 4);
  float*    norms   = (float*)carve(2 * BB * 4);

  const int TB = 256;

  // weight conversion (transposed f16)
  {
    int tw1 = LL * DD * HH;
    cvt_wT<<<cdiv(tw1, TB), TB, 0, stream>>>(enc_W1, w1hT, LL, DD, HH);
    cvt_wT<<<cdiv(tw1, TB), TB, 0, stream>>>(con_W1, w1hT + (size_t)LL * HH * DD, LL, DD, HH);
    int tw2 = LL * HH * DD;
    cvt_wT<<<cdiv(tw2, TB), TB, 0, stream>>>(enc_W2, w2hT, LL, HH, DD);
    cvt_wT<<<cdiv(tw2, TB), TB, 0, stream>>>(con_W2, w2hT + (size_t)LL * DD * HH, LL, HH, DD);
  }

  // masked input x
  hipMemcpyAsync(buf_x, feat, (size_t)NN * DD * 4, hipMemcpyDeviceToDevice, stream);
  apply_mask<<<cdiv((long long)NMASK * DD, TB), TB, 0, stream>>>(buf_x, mask_nodes, mask_token);

  // pooled accumulators
  hipMemsetAsync(pooled_g, 0, (size_t)BB * LL * DD * 4, stream);
  hipMemsetAsync(pooled_c, 0, (size_t)BB * LL * DD * 4, stream);

  // online encoder on masked x -> gh
  run_encoder(buf_x, src, dst, gid,
              w1hT, enc_g1, enc_b1, w2hT, enc_g2, enc_b2,
              pooled_g, buf_h, buf_agg, buf_z1, ah16, z1h,
              sums, sqs, cA, cB, stream);
  // contrast encoder on raw feat -> ch
  run_encoder(feat, src, dst, gid,
              w1hT + (size_t)LL * HH * DD, con_g1, con_b1,
              w2hT + (size_t)LL * DD * HH, con_g2, con_b2,
              pooled_c, buf_h, buf_agg, buf_z1, ah16, z1h,
              sums, sqs, cA, cB, stream);

  // projection head: c_h = proj(ch), c_m = proj(gh)
  proj1_k<<<cdiv(BB * 192, TB), TB, 0, stream>>>(pooled_c, pW1, pb1, hid_h);
  proj2_k<<<cdiv(BB * 128, TB), TB, 0, stream>>>(hid_h, pW2, pb2, c_h);
  proj1_k<<<cdiv(BB * 192, TB), TB, 0, stream>>>(pooled_g, pW1, pb1, hid_m);
  proj2_k<<<cdiv(BB * 128, TB), TB, 0, stream>>>(hid_m, pW2, pb2, c_m);

  // NT-Xent loss
  row_norms<<<2 * BB, 128, 0, stream>>>(c_h, c_m, norms);
  hipMemsetAsync(out, 0, sizeof(float), stream);
  nt_xent<<<BB, 256, 0, stream>>>(c_h, c_m, norms, out);
}
